// OfficialMambaModel_9354438771060
// MI455X (gfx1250) — matrix operand order repaired and checked by reading, not compiled, not run on hardware
//
#include <hip/hip_runtime.h>
#include <hip/hip_bf16.h>

typedef _Float16 half16 __attribute__((ext_vector_type(16)));
typedef _Float16 half4  __attribute__((ext_vector_type(4)));
typedef float    float8 __attribute__((ext_vector_type(8)));

#define LSEQ    1024
#define DMODEL  1024
#define DINNER  1024
#define DSTATE  32
#define KCONV   4
#define DTRANK  64
#define NLAYERS 8
#define VOCABSZ 16384

// ---------------------------------------------------------------------------
// Tiled WMMA GEMM: C[M,N] = A[M,K] * W[N,K]^T (+ optional accum), fp32 in/out,
// f16 staged in LDS, f32 accumulate via v_wmma_f32_16x16x32_f16.
// Block = 256 threads = 8 wave32 waves; tile BM=128 x BN=64, BK=64.
// Wave (wm in 0..3, wn in 0..1) computes a 32x32 output slab:
// rows [32*wm, +32), cols [32*wn, +32) -> 4 accumulators, 8 WMMAs / K-tile,
// each A/B fragment reused twice.
// ---------------------------------------------------------------------------
__global__ __launch_bounds__(256) void k_gemm_f16wmma(
    const float* __restrict__ A, int lda,
    const float* __restrict__ W, int ldw,
    const float* __restrict__ Cin,          // nullable: added to product
    float* __restrict__ C, int ldc,
    int Kdim)
{
  // stride 80 halves = 160 bytes (multiple of 16B -> aligned b64/b128 LDS ops)
  __shared__ alignas(16) _Float16 As[128][80];
  __shared__ alignas(16) _Float16 Bs[64][80];

  const int tid  = threadIdx.x;
  const int lane = tid & 31;
  const int wave = tid >> 5;
  const int wm   = wave & 3;                // 32-row slab
  const int wn   = wave >> 2;               // 32-col slab
  const int mblk = blockIdx.y * 128;
  const int nblk = blockIdx.x * 64;

  float8 acc00 = {}, acc01 = {}, acc10 = {}, acc11 = {};

  for (int kk = 0; kk < Kdim; kk += 64) {
    // ---- stage A(128x64) and W(64x64) f32 tiles into LDS as f16 ----
    float4 areg[8], breg[4];
    #pragma unroll
    for (int i = 0; i < 8; ++i) {           // A: 2048 float4, 8 per thread
      int vi = i * 256 + tid;
      int r  = vi >> 4, cv = vi & 15;       // 16 float4 per 64-wide row
      areg[i] = *(const float4*)&A[(size_t)(mblk + r) * lda + kk + cv * 4];
    }
    #pragma unroll
    for (int i = 0; i < 4; ++i) {           // B: 1024 float4, 4 per thread
      int vi = i * 256 + tid;
      int r  = vi >> 4, cv = vi & 15;
      breg[i] = *(const float4*)&W[(size_t)(nblk + r) * ldw + kk + cv * 4];
    }
    if (kk + 64 < Kdim) {                   // prefetch next K tile
      __builtin_prefetch(&A[(size_t)(mblk + (tid >> 1)) * lda + kk + 64 + ((tid & 1) << 5)], 0, 3);
      __builtin_prefetch(&W[(size_t)(nblk + (tid >> 2)) * ldw + kk + 64 + ((tid & 3) << 4)], 0, 3);
    }
    #pragma unroll
    for (int i = 0; i < 8; ++i) {
      int vi = i * 256 + tid;
      int r  = vi >> 4, cv = vi & 15;
      half4 ah = { (_Float16)areg[i].x, (_Float16)areg[i].y,
                   (_Float16)areg[i].z, (_Float16)areg[i].w };
      *(half4*)&As[r][cv * 4] = ah;         // ds_store_b64
    }
    #pragma unroll
    for (int i = 0; i < 4; ++i) {
      int vi = i * 256 + tid;
      int r  = vi >> 4, cv = vi & 15;
      half4 bh = { (_Float16)breg[i].x, (_Float16)breg[i].y,
                   (_Float16)breg[i].z, (_Float16)breg[i].w };
      *(half4*)&Bs[r][cv * 4] = bh;
    }
    __syncthreads();

    // ---- two K=32 WMMA sub-steps, 4 WMMAs each ----
    #pragma unroll
    for (int ks = 0; ks < 64; ks += 32) {
      // A fragments (16x32 f16): lanes 0-15 row M=lane, K=0..7 & 16..23;
      // lanes 16-31 same rows, K=8..15 & 24..31.
      half16 af0, af1;
      {
        int r  = 32 * wm + (lane & 15);
        int kb = ks + (lane >> 4) * 8;
        #pragma unroll
        for (int j = 0; j < 8; ++j) {
          af0[j]     = As[r][kb + j];
          af0[8 + j] = As[r][kb + 16 + j];
          af1[j]     = As[r + 16][kb + j];
          af1[8 + j] = As[r + 16][kb + 16 + j];
        }
      }
      // B fragments (32x16 f16): lane = column N; lanes 0-15 K=0..7 & 16..23;
      // lanes 16-31 K=8..15 & 24..31.
      half16 bf0, bf1;
      {
        int kb = ks + (lane >> 4) * 8;
        int n0 = 32 * wn + (lane & 15);
        #pragma unroll
        for (int j = 0; j < 8; ++j) {
          bf0[j]     = Bs[n0][kb + j];
          bf0[8 + j] = Bs[n0][kb + 16 + j];
          bf1[j]     = Bs[n0 + 16][kb + j];
          bf1[8 + j] = Bs[n0 + 16][kb + 16 + j];
        }
      }
      acc00 = __builtin_amdgcn_wmma_f32_16x16x32_f16(false, af0, false, bf0,
                                                     (short)0, acc00, false, false);
      acc01 = __builtin_amdgcn_wmma_f32_16x16x32_f16(false, af0, false, bf1,
                                                     (short)0, acc01, false, false);
      acc10 = __builtin_amdgcn_wmma_f32_16x16x32_f16(false, af1, false, bf0,
                                                     (short)0, acc10, false, false);
      acc11 = __builtin_amdgcn_wmma_f32_16x16x32_f16(false, af1, false, bf1,
                                                     (short)0, acc11, false, false);
    }
    __syncthreads();
  }

  // C/D layout: VGPR v -> M = v (lanes 0-15) or v+8 (lanes 16-31); N = lane&15.
  {
    const int c0 = nblk + 32 * wn + (lane & 15);
    const int r0 = mblk + 32 * wm + ((lane >> 4) * 8);
    #pragma unroll
    for (int v = 0; v < 8; ++v) {
      float v00 = acc00[v], v01 = acc01[v], v10 = acc10[v], v11 = acc11[v];
      if (Cin) {
        v00 += Cin[(size_t)(r0 + v) * ldc + c0];
        v01 += Cin[(size_t)(r0 + v) * ldc + c0 + 16];
        v10 += Cin[(size_t)(r0 + 16 + v) * ldc + c0];
        v11 += Cin[(size_t)(r0 + 16 + v) * ldc + c0 + 16];
      }
      C[(size_t)(r0 + v) * ldc + c0]           = v00;
      C[(size_t)(r0 + v) * ldc + c0 + 16]      = v01;
      C[(size_t)(r0 + 16 + v) * ldc + c0]      = v10;
      C[(size_t)(r0 + 16 + v) * ldc + c0 + 16] = v11;
    }
  }
}

// ---------------------------------------------------------------------------
// Embedding gather: seq[l][:] = embedding[tok[l]][:]
// ---------------------------------------------------------------------------
__global__ __launch_bounds__(256) void k_embed(
    const int* __restrict__ tok, const float* __restrict__ emb,
    float* __restrict__ seq)
{
  int l = blockIdx.x;
  int t = tok[l];
  const float4* src = (const float4*)&emb[(size_t)t * DMODEL];
  float4* dst = (float4*)&seq[(size_t)l * DMODEL];
  for (int i = threadIdx.x; i < DMODEL / 4; i += 256)
    dst[i] = src[i];
}

// ---------------------------------------------------------------------------
// RMSNorm per row (eps = 1e-5)
// ---------------------------------------------------------------------------
__global__ __launch_bounds__(256) void k_rmsnorm(
    const float* __restrict__ x, float* __restrict__ o)
{
  __shared__ float red[256];
  int l = blockIdx.x;
  float s = 0.f;
  for (int i = threadIdx.x; i < DMODEL; i += 256) {
    float v = x[l * DMODEL + i];
    s += v * v;
  }
  red[threadIdx.x] = s;
  __syncthreads();
  for (int off = 128; off > 0; off >>= 1) {
    if (threadIdx.x < off) red[threadIdx.x] += red[threadIdx.x + off];
    __syncthreads();
  }
  float r = rsqrtf(red[0] * (1.0f / DMODEL) + 1e-5f);
  for (int i = threadIdx.x; i < DMODEL; i += 256)
    o[l * DMODEL + i] = x[l * DMODEL + i] * r;
}

// ---------------------------------------------------------------------------
// Causal depthwise conv (K=4) + bias + SiLU. xi = xz[:, 0:1024].
// ---------------------------------------------------------------------------
__global__ __launch_bounds__(256) void k_conv_silu(
    const float* __restrict__ xz, const float* __restrict__ cw,
    const float* __restrict__ cb, float* __restrict__ u)
{
  int idx = blockIdx.x * 256 + threadIdx.x;   // l*1024 + d
  int l = idx >> 10, d = idx & 1023;
  float acc = cb[d];
  #pragma unroll
  for (int j = 0; j < KCONV; ++j) {
    int ls = l - (KCONV - 1) + j;
    if (ls >= 0) acc += xz[ls * (2 * DINNER) + d] * cw[d * KCONV + j];
  }
  u[idx] = acc / (1.f + __expf(-acc));        // silu
}

// ---------------------------------------------------------------------------
// dt = softplus(dt_lin + bias), in place
// ---------------------------------------------------------------------------
__global__ __launch_bounds__(256) void k_softplus(
    float* __restrict__ dt, const float* __restrict__ b)
{
  int idx = blockIdx.x * 256 + threadIdx.x;
  int d = idx & 1023;
  float x = dt[idx] + b[d];
  dt[idx] = (x > 20.f) ? x : log1pf(__expf(x));
}

// ---------------------------------------------------------------------------
// Selective scan: one thread per channel d, h[32] in registers, serial in l.
// ---------------------------------------------------------------------------
__global__ __launch_bounds__(256) void k_scan(
    const float* __restrict__ u, const float* __restrict__ dt,
    const float* __restrict__ xdbl, const float* __restrict__ A_log,
    float* __restrict__ y)
{
  int d = blockIdx.x * 256 + threadIdx.x;
  float Arow[DSTATE], h[DSTATE];
  #pragma unroll
  for (int n = 0; n < DSTATE; ++n) {
    Arow[n] = -__expf(A_log[d * DSTATE + n]);
    h[n] = 0.f;
  }
  for (int l = 0; l < LSEQ; ++l) {
    float dtt = dt[l * DINNER + d];
    float du  = dtt * u[l * DINNER + d];
    const float* Bp = &xdbl[l * (DTRANK + 2 * DSTATE) + DTRANK];
    const float* Cp = Bp + DSTATE;
    float yacc = 0.f;
    #pragma unroll
    for (int n = 0; n < DSTATE; ++n) {
      h[n] = __expf(dtt * Arow[n]) * h[n] + du * Bp[n];
      yacc += h[n] * Cp[n];
    }
    y[l * DINNER + d] = yacc;
  }
}

// ---------------------------------------------------------------------------
// y = (y + u*D) * silu(z); z = xz[:, 1024:2048]
// ---------------------------------------------------------------------------
__global__ __launch_bounds__(256) void k_gate(
    float* __restrict__ y, const float* __restrict__ u,
    const float* __restrict__ Dw, const float* __restrict__ xz)
{
  int idx = blockIdx.x * 256 + threadIdx.x;
  int l = idx >> 10, d = idx & 1023;
  float z = xz[l * (2 * DINNER) + DINNER + d];
  float v = y[idx] + u[idx] * Dw[d];
  y[idx] = v * (z / (1.f + __expf(-z)));
}

// ---------------------------------------------------------------------------
extern "C" void kernel_launch(void* const* d_in, const int* in_sizes, int n_in,
                              void* d_out, int out_size, void* d_ws, size_t ws_size,
                              hipStream_t stream) {
  const int*   tok       = (const int*)  d_in[0];
  const float* embedding = (const float*)d_in[1];
  const float* in_proj_w = (const float*)d_in[2];
  const float* conv_w    = (const float*)d_in[3];
  const float* conv_b    = (const float*)d_in[4];
  const float* x_proj_w  = (const float*)d_in[5];
  const float* dt_proj_w = (const float*)d_in[6];
  const float* dt_proj_b = (const float*)d_in[7];
  const float* A_log     = (const float*)d_in[8];
  const float* Dw        = (const float*)d_in[9];
  const float* out_proj_w= (const float*)d_in[10];
  const float* head_w    = (const float*)d_in[11];
  float* out = (float*)d_out;

  // Workspace carve-up (all fully rewritten each call)
  float* p    = (float*)d_ws;
  float* seq  = p; p += LSEQ * DMODEL;            // residual stream
  float* xn   = p; p += LSEQ * DMODEL;            // rmsnorm output
  float* xz   = p; p += LSEQ * 2 * DINNER;        // in_proj output
  float* u    = p; p += LSEQ * DINNER;            // conv+silu output
  float* xdbl = p; p += LSEQ * (DTRANK + 2 * DSTATE);
  float* dt   = p; p += LSEQ * DINNER;
  float* y    = p; p += LSEQ * DINNER;

  dim3 blk(256);

  k_embed<<<dim3(LSEQ), blk, 0, stream>>>(tok, embedding, seq);

  for (int i = 0; i < NLAYERS; ++i) {
    const float* in_w  = in_proj_w  + (size_t)i * 2 * DINNER * DMODEL;
    const float* cw    = conv_w     + (size_t)i * DINNER * KCONV;
    const float* cb    = conv_b     + (size_t)i * DINNER;
    const float* xp_w  = x_proj_w   + (size_t)i * (DTRANK + 2 * DSTATE) * DINNER;
    const float* dtp_w = dt_proj_w  + (size_t)i * DINNER * DTRANK;
    const float* dtp_b = dt_proj_b  + (size_t)i * DINNER;
    const float* Al    = A_log      + (size_t)i * DINNER * DSTATE;
    const float* Dl    = Dw         + (size_t)i * DINNER;
    const float* out_w = out_proj_w + (size_t)i * DMODEL * DINNER;

    k_rmsnorm<<<dim3(LSEQ), blk, 0, stream>>>(seq, xn);

    // xz[L, 2048] = xn[L,1024] @ in_w[2048,1024]^T
    k_gemm_f16wmma<<<dim3(2 * DINNER / 64, LSEQ / 128), blk, 0, stream>>>(
        xn, DMODEL, in_w, DMODEL, nullptr, xz, 2 * DINNER, DMODEL);

    k_conv_silu<<<dim3(LSEQ * DINNER / 256), blk, 0, stream>>>(xz, cw, cb, u);

    // xdbl[L, 128] = u @ xp_w[128,1024]^T
    k_gemm_f16wmma<<<dim3((DTRANK + 2 * DSTATE) / 64, LSEQ / 128), blk, 0, stream>>>(
        u, DINNER, xp_w, DINNER, nullptr, xdbl, DTRANK + 2 * DSTATE, DINNER);

    // dt_lin[L, 1024] = xdbl[:, :64] @ dtp_w[1024,64]^T
    k_gemm_f16wmma<<<dim3(DINNER / 64, LSEQ / 128), blk, 0, stream>>>(
        xdbl, DTRANK + 2 * DSTATE, dtp_w, DTRANK, nullptr, dt, DINNER, DTRANK);

    k_softplus<<<dim3(LSEQ * DINNER / 256), blk, 0, stream>>>(dt, dtp_b);

    k_scan<<<dim3(DINNER / 256), blk, 0, stream>>>(u, dt, xdbl, Al, y);

    k_gate<<<dim3(LSEQ * DINNER / 256), blk, 0, stream>>>(y, u, Dl, xz);

    // seq = y @ out_w[1024,1024]^T + seq   (residual fused via accum path)
    k_gemm_f16wmma<<<dim3(DMODEL / 64, LSEQ / 128), blk, 0, stream>>>(
        y, DINNER, out_w, DINNER, seq, seq, DMODEL, DINNER);
  }

  // logits[L, 16384] = seq @ head_w[16384,1024]^T
  k_gemm_f16wmma<<<dim3(VOCABSZ / 64, LSEQ / 128), blk, 0, stream>>>(
      seq, DMODEL, head_w, DMODEL, nullptr, out, VOCABSZ, DMODEL);
}
